// MultiHeadAttention_73478300500052
// MI455X (gfx1250) — matrix operand order repaired and checked by reading, not compiled, not run on hardware
//
#include <hip/hip_runtime.h>
#include <hip/hip_bf16.h>
#include <math.h>

// ---------------------------------------------------------------------------
// Problem constants (from reference)
// ---------------------------------------------------------------------------
constexpr int BATCH = 2;
constexpr int SEQ   = 2048;
constexpr int DMODEL = 1024;
constexpr int NHEADS = 16;
constexpr int HDIM   = 64;           // head dim
constexpr int MTOT   = BATCH * SEQ;  // 4096 rows for the big GEMMs

typedef __attribute__((ext_vector_type(16))) _Float16 v16h;
typedef __attribute__((ext_vector_type(8)))  _Float16 v8h;
typedef __attribute__((ext_vector_type(8)))  float    v8f;

// ---------------------------------------------------------------------------
// Fragment helpers for V_WMMA_F32_16X16X32_F16 (wave32)
//
// A (16x32 f16, row major src, ld in elements):
//   lane L<16  : row M=L,    elems 0..7 = K[0..7],  elems 8..15 = K[16..23]
//   lane L>=16 : row M=L-16, elems 0..7 = K[8..15], elems 8..15 = K[24..31]
// B (32x16 f16) loaded from a TRANSPOSED source BT[N][K] (row major, ld):
//   lane L<16  : col N=L,    elems 0..7 = K[0..7],  elems 8..15 = K[16..23]
//   lane L>=16 : col N=L-16, elems 0..7 = K[8..15], elems 8..15 = K[24..31]
// C/D (16x16 f32): elem v -> row M = (lane>>4)*8 + v, col N = lane&15
// ---------------------------------------------------------------------------
__device__ __forceinline__ v16h load_a_f16(const _Float16* __restrict__ base,
                                           int row, int ld, int k0, int lane) {
  const int r  = lane & 15;
  const int hi = lane >> 4;
  const _Float16* p = base + (size_t)(row + r) * ld + k0 + hi * 8;
  v8h lo = *(const v8h*)(p);
  v8h up = *(const v8h*)(p + 16);
  v16h a;
#pragma unroll
  for (int i = 0; i < 8; ++i) { a[i] = lo[i]; a[i + 8] = up[i]; }
  return a;
}

__device__ __forceinline__ v16h load_a_f32(const float* __restrict__ base,
                                           int row, int ld, int k0, int lane) {
  const int r  = lane & 15;
  const int hi = lane >> 4;
  const float* p = base + (size_t)(row + r) * ld + k0 + hi * 8;
  float4 x0 = *(const float4*)(p);
  float4 x1 = *(const float4*)(p + 4);
  float4 x2 = *(const float4*)(p + 16);
  float4 x3 = *(const float4*)(p + 20);
  v16h a;
  a[0]=(_Float16)x0.x; a[1]=(_Float16)x0.y; a[2]=(_Float16)x0.z; a[3]=(_Float16)x0.w;
  a[4]=(_Float16)x1.x; a[5]=(_Float16)x1.y; a[6]=(_Float16)x1.z; a[7]=(_Float16)x1.w;
  a[8]=(_Float16)x2.x; a[9]=(_Float16)x2.y; a[10]=(_Float16)x2.z; a[11]=(_Float16)x2.w;
  a[12]=(_Float16)x3.x; a[13]=(_Float16)x3.y; a[14]=(_Float16)x3.z; a[15]=(_Float16)x3.w;
  return a;
}

__device__ __forceinline__ v16h load_b_tr(const _Float16* __restrict__ bt,
                                          int n0, int ld, int k0, int lane) {
  const int r  = lane & 15;
  const int hi = lane >> 4;
  const _Float16* p = bt + (size_t)(n0 + r) * ld + k0 + hi * 8;
  v8h lo = *(const v8h*)(p);
  v8h up = *(const v8h*)(p + 16);
  v16h b;
#pragma unroll
  for (int i = 0; i < 8; ++i) { b[i] = lo[i]; b[i + 8] = up[i]; }
  return b;
}

__device__ __forceinline__ v8f wmma_f16(v16h a, v16h b, v8f c) {
  return __builtin_amdgcn_wmma_f32_16x16x32_f16(false, a, false, b, (short)0, c,
                                                false, false);
}

// ---------------------------------------------------------------------------
// Kernel: f32 -> f16 elementwise convert (vector x4)
// ---------------------------------------------------------------------------
__global__ __launch_bounds__(256) void cvt_f16_kernel(const float* __restrict__ in,
                                                      _Float16* __restrict__ out,
                                                      int n) {
  int i = (blockIdx.x * 256 + threadIdx.x) * 4;
  if (i + 3 < n) {
    float4 v = *(const float4*)(in + i);
    out[i + 0] = (_Float16)v.x;
    out[i + 1] = (_Float16)v.y;
    out[i + 2] = (_Float16)v.z;
    out[i + 3] = (_Float16)v.w;
  }
}

// ---------------------------------------------------------------------------
// Kernel: transpose + convert weight  in[K=1024][N=1024] f32 -> out[N][K] f16
// ---------------------------------------------------------------------------
__global__ __launch_bounds__(256) void transpose_cvt_kernel(const float* __restrict__ in,
                                                            _Float16* __restrict__ out) {
  __shared__ _Float16 tile[32][33];
  const int bx = blockIdx.x * 32;  // n base
  const int by = blockIdx.y * 32;  // k base
  const int tx = threadIdx.x & 31;
  const int ty = threadIdx.x >> 5;  // 0..7
#pragma unroll
  for (int i = 0; i < 32; i += 8)
    tile[ty + i][tx] = (_Float16)in[(size_t)(by + ty + i) * DMODEL + bx + tx];
  __syncthreads();
#pragma unroll
  for (int i = 0; i < 32; i += 8)
    out[(size_t)(bx + ty + i) * DMODEL + by + tx] = tile[tx][ty + i];
}

// ---------------------------------------------------------------------------
// Kernel: projection GEMM   C[M=4096,N=1024] = A[M,K=1024] * BT[N,K]^T + bias
//   mode 0: store f16 head-split [B,H,S,Dh]      (Q, K projections)
//   mode 1: store f16 head-split transposed [B,H,Dh,S]   (V projection)
//   mode 2: store f32 row-major [M,N]            (output projection)
// Block tile 128x128, 8 waves (4x2), wave tile 32x64 -> 2x4 WMMA accumulators.
// ---------------------------------------------------------------------------
__global__ __launch_bounds__(256) void proj_kernel(const _Float16* __restrict__ A,
                                                   const _Float16* __restrict__ BT,
                                                   const float* __restrict__ bias,
                                                   _Float16* __restrict__ out_h,
                                                   float* __restrict__ out_f,
                                                   int mode) {
  const int lane = threadIdx.x & 31;
  const int wave = threadIdx.x >> 5;
  const int wm = wave >> 1;  // 0..3
  const int wn = wave & 1;   // 0..1
  const int m_base = blockIdx.x * 128 + wm * 32;
  const int n_base = blockIdx.y * 128 + wn * 64;
  const int r  = lane & 15;
  const int hi = lane >> 4;

  v8f acc[2][4] = {};
  for (int kk = 0; kk < DMODEL; kk += 32) {
    if (kk + 32 < DMODEL) {
      __builtin_prefetch(A + (size_t)(m_base + r) * DMODEL + kk + 32, 0, 3);
      __builtin_prefetch(BT + (size_t)(n_base + r) * DMODEL + kk + 32, 0, 3);
    }
    v16h a0 = load_a_f16(A, m_base, DMODEL, kk, lane);
    v16h a1 = load_a_f16(A, m_base + 16, DMODEL, kk, lane);
#pragma unroll
    for (int ni = 0; ni < 4; ++ni) {
      v16h b = load_b_tr(BT, n_base + ni * 16, DMODEL, kk, lane);
      acc[0][ni] = wmma_f16(a0, b, acc[0][ni]);
      acc[1][ni] = wmma_f16(a1, b, acc[1][ni]);
    }
  }

#pragma unroll
  for (int mi = 0; mi < 2; ++mi) {
#pragma unroll
    for (int ni = 0; ni < 4; ++ni) {
      const int n = n_base + ni * 16 + r;
      const float bv = bias[n];
#pragma unroll
      for (int v = 0; v < 8; ++v) {
        const int m = m_base + mi * 16 + hi * 8 + v;
        const float val = acc[mi][ni][v] + bv;
        if (mode == 2) {
          out_f[(size_t)m * DMODEL + n] = val;
        } else if (mode == 0) {
          // [B,H,S,Dh]
          out_h[((size_t)((m >> 11) * NHEADS + (n >> 6)) * SEQ + (m & 2047)) * HDIM
                + (n & 63)] = (_Float16)val;
        } else {
          // [B,H,Dh,S]
          out_h[((size_t)((m >> 11) * NHEADS + (n >> 6)) * HDIM + (n & 63)) * SEQ
                + (m & 2047)] = (_Float16)val;
        }
      }
    }
  }
}

// ---------------------------------------------------------------------------
// Kernel: scores  S[bh][q][k] = (Q[bh][q][:] . K[bh][k][:]) / sqrt(Dh)
// Block tile 64(q) x 128(k), 8 waves (4x2), wave tile 16x64.
// Qh, Kh in [B,H,S,Dh]; Kh rows act directly as the transposed-B source.
// ---------------------------------------------------------------------------
__global__ __launch_bounds__(256) void scores_kernel(const _Float16* __restrict__ Qh,
                                                     const _Float16* __restrict__ Kh,
                                                     float* __restrict__ attn) {
  const int lane = threadIdx.x & 31;
  const int wave = threadIdx.x >> 5;
  const int wm = wave >> 1;  // 0..3
  const int wn = wave & 1;   // 0..1
  const int q0 = blockIdx.x * 64 + wm * 16;
  const int n0base = blockIdx.y * 128 + wn * 64;
  const int bh = blockIdx.z;
  const _Float16* Qb = Qh + (size_t)bh * SEQ * HDIM;
  const _Float16* Kb = Kh + (size_t)bh * SEQ * HDIM;
  const int r  = lane & 15;
  const int hi = lane >> 4;

  v8f acc[4] = {};
#pragma unroll
  for (int kk = 0; kk < HDIM; kk += 32) {
    v16h a = load_a_f16(Qb, q0, HDIM, kk, lane);
#pragma unroll
    for (int ns = 0; ns < 4; ++ns) {
      v16h b = load_b_tr(Kb, n0base + ns * 16, HDIM, kk, lane);
      acc[ns] = wmma_f16(a, b, acc[ns]);
    }
  }

  float* ab = attn + (size_t)bh * SEQ * SEQ;
  const float scale = 0.125f;  // 1/sqrt(64)
#pragma unroll
  for (int ns = 0; ns < 4; ++ns) {
    const int col = n0base + ns * 16 + r;
#pragma unroll
    for (int v = 0; v < 8; ++v) {
      const int row = q0 + hi * 8 + v;
      ab[(size_t)row * SEQ + col] = acc[ns][v] * scale;
    }
  }
}

// ---------------------------------------------------------------------------
// Kernel: row softmax over 2048 elements, in place. One block (256 thr) / row.
// ---------------------------------------------------------------------------
__global__ __launch_bounds__(256) void softmax_kernel(float* __restrict__ attn) {
  float* p = attn + (size_t)blockIdx.x * SEQ;
  const int t = threadIdx.x;
  __shared__ float red[256];

  float vals[8];
  float mx = -INFINITY;
#pragma unroll
  for (int i = 0; i < 8; ++i) {
    vals[i] = p[t + i * 256];
    mx = fmaxf(mx, vals[i]);
  }
  red[t] = mx;
  __syncthreads();
  for (int s = 128; s > 0; s >>= 1) {
    if (t < s) red[t] = fmaxf(red[t], red[t + s]);
    __syncthreads();
  }
  mx = red[0];
  __syncthreads();

  float sum = 0.f;
#pragma unroll
  for (int i = 0; i < 8; ++i) {
    vals[i] = __expf(vals[i] - mx);
    sum += vals[i];
  }
  red[t] = sum;
  __syncthreads();
  for (int s = 128; s > 0; s >>= 1) {
    if (t < s) red[t] += red[t + s];
    __syncthreads();
  }
  const float inv = 1.0f / red[0];
#pragma unroll
  for (int i = 0; i < 8; ++i) p[t + i * 256] = vals[i] * inv;
}

// ---------------------------------------------------------------------------
// Kernel: context = P @ V  per head.  P [S,S] f32 (converted to f16 on load),
// Vt [B,H,Dh,S] f16 (transposed-B source).  Block 128 q-rows x 64 dh, 8 waves.
// Output ctx f16 in [B, S, H*Dh] = [4096, 1024] row-major for final GEMM.
// ---------------------------------------------------------------------------
__global__ __launch_bounds__(256) void pv_kernel(const float* __restrict__ attn,
                                                 const _Float16* __restrict__ Vt,
                                                 _Float16* __restrict__ ctx) {
  const int lane = threadIdx.x & 31;
  const int wave = threadIdx.x >> 5;
  const int q0 = blockIdx.x * 128 + wave * 16;
  const int bh = blockIdx.y;
  const float* Pb = attn + (size_t)bh * SEQ * SEQ;
  const _Float16* Vb = Vt + (size_t)bh * HDIM * SEQ;
  const int r  = lane & 15;
  const int hi = lane >> 4;

  v8f acc[4] = {};
  for (int kk = 0; kk < SEQ; kk += 32) {
    v16h a = load_a_f32(Pb, q0, SEQ, kk, lane);
#pragma unroll
    for (int ns = 0; ns < 4; ++ns) {
      v16h b = load_b_tr(Vb, ns * 16, SEQ, kk, lane);
      acc[ns] = wmma_f16(a, b, acc[ns]);
    }
  }

  const int b = bh >> 4;
  const int h = bh & 15;
#pragma unroll
  for (int ns = 0; ns < 4; ++ns) {
    const int dh = ns * 16 + r;
#pragma unroll
    for (int v = 0; v < 8; ++v) {
      const int s = q0 + hi * 8 + v;
      ctx[((size_t)(b * SEQ + s)) * DMODEL + h * HDIM + dh] = (_Float16)acc[ns][v];
    }
  }
}

// ---------------------------------------------------------------------------
// Host launcher
// ---------------------------------------------------------------------------
extern "C" void kernel_launch(void* const* d_in, const int* in_sizes, int n_in,
                              void* d_out, int out_size, void* d_ws, size_t ws_size,
                              hipStream_t stream) {
  const float* q  = (const float*)d_in[0];
  const float* k  = (const float*)d_in[1];
  const float* v  = (const float*)d_in[2];
  const float* Wq = (const float*)d_in[3];
  const float* bq = (const float*)d_in[4];
  const float* Wk = (const float*)d_in[5];
  const float* bk = (const float*)d_in[6];
  const float* Wv = (const float*)d_in[7];
  const float* bv = (const float*)d_in[8];
  const float* Wo = (const float*)d_in[9];
  const float* bo = (const float*)d_in[10];

  float* out  = (float*)d_out;                       // [4096, 1024]
  float* attn = out + (size_t)MTOT * DMODEL;         // [2,16,2048,2048]

  // Workspace layout (halfs), total 64 MiB
  const size_t NXY = (size_t)MTOT * DMODEL;          // 4 Mi elements
  const size_t NW  = (size_t)DMODEL * DMODEL;        // 1 Mi elements
  _Float16* w   = (_Float16*)d_ws;
  _Float16* Xq  = w;
  _Float16* Xk  = Xq + NXY;
  _Float16* Xv  = Xk + NXY;
  _Float16* WqT = Xv + NXY;
  _Float16* WkT = WqT + NW;
  _Float16* WvT = WkT + NW;
  _Float16* WoT = WvT + NW;
  _Float16* Qh  = WoT + NW;   // [B,H,S,Dh]
  _Float16* Kh  = Qh + NXY;   // [B,H,S,Dh]
  _Float16* Vt  = Kh + NXY;   // [B,H,Dh,S]
  _Float16* ctx = Vt + NXY;   // [B,S,D]
  (void)ws_size; (void)n_in; (void)in_sizes; (void)out_size;

  // 1) convert inputs to f16
  {
    const int n = (int)NXY;
    const int grid = n / (256 * 4);
    cvt_f16_kernel<<<grid, 256, 0, stream>>>(q, Xq, n);
    cvt_f16_kernel<<<grid, 256, 0, stream>>>(k, Xk, n);
    cvt_f16_kernel<<<grid, 256, 0, stream>>>(v, Xv, n);
  }

  // 2) transpose + convert weights
  {
    dim3 g(DMODEL / 32, DMODEL / 32);
    transpose_cvt_kernel<<<g, 256, 0, stream>>>(Wq, WqT);
    transpose_cvt_kernel<<<g, 256, 0, stream>>>(Wk, WkT);
    transpose_cvt_kernel<<<g, 256, 0, stream>>>(Wv, WvT);
    transpose_cvt_kernel<<<g, 256, 0, stream>>>(Wo, WoT);
  }

  // 3) Q/K/V projections (WMMA)
  {
    dim3 g(MTOT / 128, DMODEL / 128);
    proj_kernel<<<g, 256, 0, stream>>>(Xq, WqT, bq, Qh, nullptr, 0);
    proj_kernel<<<g, 256, 0, stream>>>(Xk, WkT, bk, Kh, nullptr, 0);
    proj_kernel<<<g, 256, 0, stream>>>(Xv, WvT, bv, Vt, nullptr, 1);
  }

  // 4) attention scores -> d_out attention region (pre-softmax, scaled)
  {
    dim3 g(SEQ / 64, SEQ / 128, BATCH * NHEADS);
    scores_kernel<<<g, 256, 0, stream>>>(Qh, Kh, attn);
  }

  // 5) softmax in place (attention probabilities are part of the output)
  softmax_kernel<<<BATCH * NHEADS * SEQ, 256, 0, stream>>>(attn);

  // 6) context = P @ V  (WMMA)
  {
    dim3 g(SEQ / 128, BATCH * NHEADS);
    pv_kernel<<<g, 256, 0, stream>>>(attn, Vt, ctx);
  }

  // 7) output projection -> d_out (f32) (WMMA)
  {
    dim3 g(MTOT / 128, DMODEL / 128);
    proj_kernel<<<g, 256, 0, stream>>>(ctx, WoT, bo, nullptr, out, 2);
  }
}
